// GatedGraphConvNetNoBatchNorm_50749333569630
// MI455X (gfx1250) — matrix operand order repaired and checked by reading, not compiled, not run on hardware
//
#include <hip/hip_runtime.h>
#include <hip/hip_bf16.h>

// ---------------------------------------------------------------------------
// GatedGraphConv layer for MI455X (gfx1250, wave32, WMMA).
//  - node-level GEMMs (U,V,A,B) + per-edge GEMM (C) via v_wmma_f32_16x16x32_f16
//  - scatter softmax: ordered-uint atomicMax + global_atomic_add_f32
//  - 192MB L2 holds all per-node scratch arrays (179MB); e_hat/e_new streams
//    use non-temporal hints so they do not thrash that working set.
//  - wave/tile IDs forced scalar via readfirstlane so tile-guards compile to
//    s_cbranch (no EXEC save/restore around stores).
// ---------------------------------------------------------------------------

typedef _Float16 half16  __attribute__((ext_vector_type(16)));
typedef float    float8v __attribute__((ext_vector_type(8)));
typedef float    float4v __attribute__((ext_vector_type(4)));

#define D_FEAT    64
#define WPB       8   // waves per 256-thread block
#define ROW_TILES 2   // 16-row tiles per wave in node_gemm (32 rows/wave)

__device__ __forceinline__ float leaky_relu(float v) {
    return v >= 0.0f ? v : 0.2f * v;
}

// order-preserving float -> uint mapping (monotone under unsigned compare)
__device__ __forceinline__ unsigned encodeOrd(float v) {
    unsigned b = __float_as_uint(v);
    return (b & 0x80000000u) ? ~b : (b | 0x80000000u);
}
__device__ __forceinline__ float decodeOrd(unsigned k) {
    unsigned b = (k & 0x80000000u) ? (k & 0x7fffffffu) : ~k;
    return __uint_as_float(b);
}

// A fragment: 16x32 f16 tile (M=rows, K=features kb..kb+31).
// Contiguous k order used here (B matches): lanes 0-15 row M=lane, halves 0-15 -> K=kb+0..15
//             lanes 16-31 row M=lane-16, K offset +16.
__device__ __forceinline__ half16 load_a_frag(const float* rowPtr, int kb, int lane) {
    const int koff = (lane >> 4) << 4;
    half16 a;
#pragma unroll
    for (int i = 0; i < 8; ++i) {
        a[i]     = (_Float16)rowPtr[kb + koff + i];
        a[8 + i] = (_Float16)rowPtr[kb + koff + 8 + i];
    }
    return a;
}

// B fragment: 32x16 f16 tile of W^T, i.e. B[k][n] = W[n][k] (row-major W[64][64]).
// lane holds column n = nb + (lane&15); halves i -> K = kb + (lane>>4)*16 + i.
__device__ __forceinline__ half16 load_b_frag(const float* __restrict__ W, int nb, int kb, int lane) {
    const int n  = nb + (lane & 15);
    const int kB = kb + ((lane >> 4) << 4);
    const float* p = W + n * D_FEAT + kB;
    half16 b;
#pragma unroll
    for (int i = 0; i < 16; ++i) b[i] = (_Float16)p[i];
    return b;
}

// ---------------------------------------------------------------------------
// 0) zero-init of [mEnc | denom | y] (zero bits == encoded -inf == 0.0f)
// ---------------------------------------------------------------------------
__global__ void init_kernel(unsigned* __restrict__ p, int n) {
    int t = blockIdx.x * blockDim.x + threadIdx.x;
    if (t < n) p[t] = 0u;
}

// ---------------------------------------------------------------------------
// 1) node GEMMs: Ux = x U^T + U_b, Vx, Ax, Bx.  One wave handles 2 row-tiles
//    (32 rows); A fragments for both tiles held in registers so each weight
//    B-fragment feeds 2 WMMAs (2x less L2 weight refetch, no spills).
//    4 weights x 4 n-tiles x 2 k-steps x 2 tiles = 64 WMMAs / wave.
// ---------------------------------------------------------------------------
__global__ void __launch_bounds__(256, 1)
node_gemm_kernel(const float* __restrict__ x,
                 const float* __restrict__ Uw, const float* __restrict__ Ub,
                 const float* __restrict__ Vw, const float* __restrict__ Vb,
                 const float* __restrict__ Aw, const float* __restrict__ Ab,
                 const float* __restrict__ Bw, const float* __restrict__ Bb,
                 float* __restrict__ Ux, float* __restrict__ Vx,
                 float* __restrict__ Ax, float* __restrict__ Bx,
                 int nRows, int nWaves) {
    // wave id is identical across all 32 lanes -> force it into an SGPR so
    // every tile-level guard below is a scalar branch (no EXEC juggling).
    const int wave = __builtin_amdgcn_readfirstlane(
        (int)((blockIdx.x * blockDim.x + threadIdx.x) >> 5));
    const int lane = threadIdx.x & 31;
    if (wave >= nWaves) return;

    const int rowbase0 = wave * (16 * ROW_TILES);

    // A fragments for the row-tiles (clamped loads; stores guarded below)
    half16 a[ROW_TILES][2];
#pragma unroll
    for (int t = 0; t < ROW_TILES; ++t) {
        int row = rowbase0 + t * 16 + (lane & 15);
        if (row >= nRows) row = nRows - 1;
        const float* rp = x + (size_t)row * D_FEAT;
        a[t][0] = load_a_frag(rp, 0,  lane);
        a[t][1] = load_a_frag(rp, 32, lane);
    }

    const float* Ws[4]   = {Uw, Vw, Aw, Bw};
    const float* bias[4] = {Ub, Vb, Ab, Bb};
    float*       out[4]  = {Ux, Vx, Ax, Bx};

    const int n0   = lane & 15;
    const int msub = (lane >> 4) << 3;

#pragma unroll
    for (int w = 0; w < 4; ++w) {
        const float* W = Ws[w];
        float* o = out[w];
#pragma unroll
        for (int nb = 0; nb < D_FEAT; nb += 16) {
            const float bval = bias[w][nb + n0];
            float8v acc[ROW_TILES];
#pragma unroll
            for (int t = 0; t < ROW_TILES; ++t)
#pragma unroll
                for (int r = 0; r < 8; ++r) acc[t][r] = bval;

            const half16 b0 = load_b_frag(W, nb, 0, lane);
#pragma unroll
            for (int t = 0; t < ROW_TILES; ++t)
                acc[t] = __builtin_amdgcn_wmma_f32_16x16x32_f16(
                    false, a[t][0], false, b0, (short)0, acc[t], false, false);

            const half16 b1 = load_b_frag(W, nb, 32, lane);
#pragma unroll
            for (int t = 0; t < ROW_TILES; ++t)
                acc[t] = __builtin_amdgcn_wmma_f32_16x16x32_f16(
                    false, a[t][1], false, b1, (short)0, acc[t], false, false);

#pragma unroll
            for (int t = 0; t < ROW_TILES; ++t) {
                const int tilebase = rowbase0 + t * 16;   // scalar
                float* op = o + (size_t)(tilebase + msub) * D_FEAT + nb + n0;
                if (tilebase + 16 <= nRows) {             // scalar branch
                    // full tile: unpredicated stores
#pragma unroll
                    for (int r = 0; r < 8; ++r) op[(size_t)r * D_FEAT] = acc[t][r];
                } else {
                    // tail tile (rare)
#pragma unroll
                    for (int r = 0; r < 8; ++r)
                        if (tilebase + msub + r < nRows) op[(size_t)r * D_FEAT] = acc[t][r];
                }
            }
        }
    }
}

// ---------------------------------------------------------------------------
// 2) segment max over src:  mEnc[src[e]][d] = max(..., enc(e_hat[e][d]))
// ---------------------------------------------------------------------------
__global__ void edge_max_kernel(const float* __restrict__ e_hat,
                                const int* __restrict__ srcIdx,
                                unsigned* __restrict__ mEnc, int total) {
    const int t = blockIdx.x * blockDim.x + threadIdx.x;
    if (t >= total) return;
    const int e = t >> 6;
    const int d = t & 63;
    const int s = srcIdx[e];
    const float v = __builtin_nontemporal_load(e_hat + t);
    atomicMax(&mEnc[s * D_FEAT + d], encodeOrd(v));
}

// ---------------------------------------------------------------------------
// 3) segment sum of exp(e_hat - m[src])
// ---------------------------------------------------------------------------
__global__ void edge_sum_kernel(const float* __restrict__ e_hat,
                                const int* __restrict__ srcIdx,
                                const unsigned* __restrict__ mEnc,
                                float* __restrict__ denom, int total) {
    const int t = blockIdx.x * blockDim.x + threadIdx.x;
    if (t >= total) return;
    const int e = t >> 6;
    const int d = t & 63;
    const int s = srcIdx[e];
    const float v  = __builtin_nontemporal_load(e_hat + t);
    const float mx = decodeOrd(mEnc[s * D_FEAT + d]);
    unsafeAtomicAdd(&denom[s * D_FEAT + d], __expf(v - mx));
}

// ---------------------------------------------------------------------------
// 4) fused edge kernel: per 16-edge tile (one wave):
//      ehC  = e_hat @ C_w^T            (8 WMMAs, e_hat staged in LDS)
//      e    = exp(e_hat - m[src]) / denom[src]
//      y[src] += Vx[dst] * e           (global_atomic_add_f32)
//      e_new = e_hat + leaky(Ax[src] + Bx[dst] + ehC + C_b)   (NT store)
// ---------------------------------------------------------------------------
__global__ void edge_fused_kernel(const float* __restrict__ e_hat,
                                  const int* __restrict__ srcIdx,
                                  const int* __restrict__ dstIdx,
                                  const float* __restrict__ Cw, const float* __restrict__ Cb,
                                  const unsigned* __restrict__ mEnc,
                                  const float* __restrict__ denom,
                                  const float* __restrict__ Vx,
                                  const float* __restrict__ Ax,
                                  const float* __restrict__ Bx,
                                  float* __restrict__ yAgg,
                                  float* __restrict__ eNew,
                                  int nEdges, int nTiles) {
    __shared__ __align__(16) float sEh[WPB][16 * D_FEAT];
    __shared__ int sIdx[WPB][32];

    const int wib  = __builtin_amdgcn_readfirstlane((int)(threadIdx.x >> 5));
    const int lane = threadIdx.x & 31;
    const int tile = __builtin_amdgcn_readfirstlane((int)(blockIdx.x * WPB + wib));
    if (tile >= nTiles) return;            // scalar branch

    const int ebase = tile * 16;           // scalar
    float* lsEh = sEh[wib];
    int*   lsIx = sIdx[wib];

    // stage 16x64 e_hat tile into LDS (256 x 128-bit, coalesced, non-temporal)
    const float4v* g4 = (const float4v*)(e_hat);
    float4v*       l4 = (float4v*)lsEh;
    const int gmax = nEdges * 16 - 1;      // #float4v in e_hat - 1
#pragma unroll
    for (int i = 0; i < 8; ++i) {
        int gi = ebase * 16 + i * 32 + lane;
        if (gi > gmax) gi = gmax;
        l4[i * 32 + lane] = __builtin_nontemporal_load(g4 + gi);
    }
    if (lane < 16) {
        int e = ebase + lane;
        if (e >= nEdges) e = nEdges - 1;
        lsIx[lane]      = srcIdx[e];
        lsIx[16 + lane] = dstIdx[e];
    }
    // intra-wave LDS RAW: compiler inserts s_wait_dscnt

    // e_hat @ C_w^T  via WMMA
    const float* rp = lsEh + (lane & 15) * D_FEAT;
    const half16 a0 = load_a_frag(rp, 0,  lane);
    const half16 a1 = load_a_frag(rp, 32, lane);

    float8v acc[4];
#pragma unroll
    for (int nt = 0; nt < 4; ++nt) {
        float8v c = {};
        const half16 b0 = load_b_frag(Cw, nt * 16, 0, lane);
        c = __builtin_amdgcn_wmma_f32_16x16x32_f16(false, a0, false, b0,
                                                   (short)0, c, false, false);
        const half16 b1 = load_b_frag(Cw, nt * 16, 32, lane);
        c = __builtin_amdgcn_wmma_f32_16x16x32_f16(false, a1, false, b1,
                                                   (short)0, c, false, false);
        acc[nt] = c;
    }

    // element-wise epilogue in the C/D layout: lane owns (m = (lane>>4)*8 + r,
    // feat = nt*16 + (lane&15)).  Node-array gathers are half-wave coalesced
    // and live in L2 (179MB working set < 192MB L2).  Guard hoisted to a
    // scalar branch so the common path is unpredicated.
    const int n0    = lane & 15;
    const int mbase = (lane >> 4) << 3;

    auto epilogue = [&](bool full) {
#pragma unroll
        for (int nt = 0; nt < 4; ++nt) {
            const int feat = nt * 16 + n0;
            const float cb = Cb[feat];
#pragma unroll
            for (int r = 0; r < 8; ++r) {
                const int m  = mbase + r;
                const int eg = ebase + m;
                if (!full && eg >= nEdges) continue;
                const int s   = lsIx[m];
                const int dd  = lsIx[16 + m];
                const float eh = lsEh[m * D_FEAT + feat];
                const int so   = s  * D_FEAT + feat;
                const int dofs = dd * D_FEAT + feat;
                const float mx  = decodeOrd(mEnc[so]);
                const float den = denom[so];
                const float e   = __expf(eh - mx) / den;
                unsafeAtomicAdd(&yAgg[so], Vx[dofs] * e);
                const float t  = Ax[so] + Bx[dofs] + acc[nt][r] + cb;
                const float en = eh + leaky_relu(t);
                __builtin_nontemporal_store(en, eNew + (size_t)eg * D_FEAT + feat);
            }
        }
    };
    if (ebase + 16 <= nEdges) epilogue(true);   // scalar: common path
    else                      epilogue(false);  // scalar: tail tile
}

// ---------------------------------------------------------------------------
// 5) node residual: x_new = x + leaky(Ux + y)
// ---------------------------------------------------------------------------
__global__ void node_update_kernel(const float* __restrict__ x,
                                   const float* __restrict__ Ux,
                                   const float* __restrict__ y,
                                   float* __restrict__ xNew, int total) {
    const int t = blockIdx.x * blockDim.x + threadIdx.x;
    if (t < total) {
        xNew[t] = x[t] + leaky_relu(Ux[t] + y[t]);
    }
}

// ---------------------------------------------------------------------------
// launch
// ---------------------------------------------------------------------------
extern "C" void kernel_launch(void* const* d_in, const int* in_sizes, int n_in,
                              void* d_out, int out_size, void* d_ws, size_t ws_size,
                              hipStream_t stream) {
    const float* x     = (const float*)d_in[0];
    const float* e_hat = (const float*)d_in[1];
    const int*   eidx  = (const int*)d_in[2];     // [2,E] flattened
    const float* Uw = (const float*)d_in[3];  const float* Ub = (const float*)d_in[4];
    const float* Vw = (const float*)d_in[5];  const float* Vb = (const float*)d_in[6];
    const float* Aw = (const float*)d_in[7];  const float* Ab = (const float*)d_in[8];
    const float* Bw = (const float*)d_in[9];  const float* Bb = (const float*)d_in[10];
    const float* Cw = (const float*)d_in[11]; const float* Cb = (const float*)d_in[12];

    const int N  = in_sizes[0] / D_FEAT;
    const int E  = in_sizes[1] / D_FEAT;
    const int ND = N * D_FEAT;
    const int ED = E * D_FEAT;

    const int* srcIdx = eidx;
    const int* dstIdx = eidx + E;

    // workspace: [mEnc | denom | y | Ux | Vx | Ax | Bx], each N*64 f32
    unsigned* mEnc  = (unsigned*)d_ws;
    float*    denom = (float*)d_ws + (size_t)ND;
    float*    yAgg  = (float*)d_ws + (size_t)2 * ND;
    float*    Ux    = (float*)d_ws + (size_t)3 * ND;
    float*    Vx    = (float*)d_ws + (size_t)4 * ND;
    float*    Ax    = (float*)d_ws + (size_t)5 * ND;
    float*    Bx    = (float*)d_ws + (size_t)6 * ND;

    float* xNew = (float*)d_out;
    float* eNew = (float*)d_out + (size_t)ND;

    const int threads = 256;

    // 0) init mEnc/denom/y (contiguous, zero bits)
    {
        const int tot = 3 * ND;
        init_kernel<<<(tot + threads - 1) / threads, threads, 0, stream>>>(mEnc, tot);
    }
    // 1) node GEMMs (one wave per 32 rows)
    {
        const int nWaves = (N + 16 * ROW_TILES - 1) / (16 * ROW_TILES);
        const int blocks = (nWaves + WPB - 1) / WPB;
        node_gemm_kernel<<<blocks, threads, 0, stream>>>(
            x, Uw, Ub, Vw, Vb, Aw, Ab, Bw, Bb, Ux, Vx, Ax, Bx, N, nWaves);
    }
    // 2) segment max
    edge_max_kernel<<<(ED + threads - 1) / threads, threads, 0, stream>>>(
        e_hat, srcIdx, mEnc, ED);
    // 3) segment sum of exp
    edge_sum_kernel<<<(ED + threads - 1) / threads, threads, 0, stream>>>(
        e_hat, srcIdx, mEnc, denom, ED);
    // 4) fused edge kernel
    {
        const int nTiles = (E + 15) / 16;
        const int blocks = (nTiles + WPB - 1) / WPB;
        edge_fused_kernel<<<blocks, threads, 0, stream>>>(
            e_hat, srcIdx, dstIdx, Cw, Cb, mEnc, denom, Vx, Ax, Bx,
            yAgg, eNew, E, nTiles);
    }
    // 5) node residual update
    node_update_kernel<<<(ND + threads - 1) / threads, threads, 0, stream>>>(
        x, Ux, yAgg, xNew, ND);
}
